// AttentionOut_68624987455542
// MI455X (gfx1250) — matrix operand order repaired and checked by reading, not compiled, not run on hardware
//
#include <hip/hip_runtime.h>
#include <hip/hip_bf16.h>

typedef _Float16 f16;
typedef __attribute__((ext_vector_type(16))) _Float16 v16h;
typedef __attribute__((ext_vector_type(8)))  _Float16 v8h;
typedef __attribute__((ext_vector_type(8)))  float    v8f;

#define S_LEN   2048
#define DMODEL  1024
#define NH      16
#define DH      64
#define QTILE   64
#define KTILE   32
#define IGNORE_VAL -100000.0f
// 1/sqrt(64) * log2(e): softmax done in exp2 domain
#define SCALE_LOG2E 0.1803368801111204f

union V16 { v16h v; v8h h[2]; };

__device__ __forceinline__ void async_b128(uint32_t lds_off, const void* gptr) {
  asm volatile("global_load_async_to_lds_b128 %0, %1, off"
               :: "v"(lds_off), "v"((unsigned long long)(uintptr_t)gptr)
               : "memory");
}
#define WAIT_ASYNC(n) asm volatile("s_wait_asynccnt " #n ::: "memory")
__device__ __forceinline__ uint32_t lds_addr(const void* p) {
  return (uint32_t)(uintptr_t)p;   // low 32 bits of LDS aperture addr = LDS offset
}

// ---------------------------------------------------------------------------
// Kernel A: k -> f16 [b][s][h*64+d]; v -> f16 transposed vt[b][h][d][s]
// ---------------------------------------------------------------------------
__global__ __launch_bounds__(256) void convert_kv_kernel(
    const float* __restrict__ kg, const float* __restrict__ vg,
    f16* __restrict__ kf, f16* __restrict__ vt)
{
  const int bid = blockIdx.x;
  const int st  = bid % (S_LEN / 64);
  const int h   = (bid / (S_LEN / 64)) % NH;
  const int b   = bid / ((S_LEN / 64) * NH);
  const int s0  = st * 64;
  const int tid = threadIdx.x;
  __shared__ __align__(16) f16 T[64][72];
#pragma unroll
  for (int i = 0; i < 16; ++i) {
    int flat = i * 256 + tid;
    int row = flat >> 6, d = flat & 63;
    size_t g = ((size_t)(b * S_LEN) + s0 + row) * DMODEL + h * DH + d;
    kf[g]     = (f16)kg[g];
    T[row][d] = (f16)vg[g];
  }
  __syncthreads();
#pragma unroll
  for (int i = 0; i < 16; ++i) {
    int flat = i * 256 + tid;
    int d = flat >> 6, ss = flat & 63;
    vt[((size_t)((b * NH + h) * DH) + d) * S_LEN + s0 + ss] = T[ss][d];
  }
}

// ---------------------------------------------------------------------------
// Kernel B: W_O [k][n] f32 -> wt [n][k] f16
// ---------------------------------------------------------------------------
__global__ __launch_bounds__(256) void convert_w_kernel(
    const float* __restrict__ W, f16* __restrict__ wt)
{
  const int n0 = blockIdx.x * 64;
  const int k0 = blockIdx.y * 64;
  const int tid = threadIdx.x;
  __shared__ __align__(16) f16 T[64][72];
#pragma unroll
  for (int i = 0; i < 16; ++i) {
    int flat = i * 256 + tid;
    int row = flat >> 6, col = flat & 63;
    T[row][col] = (f16)W[(size_t)(k0 + row) * DMODEL + n0 + col];
  }
  __syncthreads();
#pragma unroll
  for (int i = 0; i < 16; ++i) {
    int flat = i * 256 + tid;
    int nn = flat >> 6, kk = flat & 63;
    wt[(size_t)(n0 + nn) * DMODEL + k0 + kk] = T[kk][nn];
  }
}

// Per-row online-softmax step, optionally with causal masking.  MASKED is a
// compile-time flag so the two variants live in separate scalar-branched code.
#define SOFTMAX_LOOP(MASKED)                                                  \
  _Pragma("unroll")                                                           \
  for (int r = 0; r < 8; ++r) {                                               \
    float s0 = c0[r] * SCALE_LOG2E;                                           \
    float s1 = c1[r] * SCALE_LOG2E;                                           \
    if (MASKED) {                                                             \
      int qgl  = q_row + r + 8 * hfl;                                         \
      int key0 = k0 + l16;                                                    \
      s0 = (key0      <= qgl) ? s0 : IGNORE_VAL;                              \
      s1 = (key0 + 16 <= qgl) ? s1 : IGNORE_VAL;                              \
    }                                                                         \
    float mx = fmaxf(s0, s1);                                                 \
    _Pragma("unroll")                                                         \
    for (int off = 1; off < 16; off <<= 1)                                    \
      mx = fmaxf(mx, __shfl_xor(mx, off, 32));                                \
    float mnew = fmaxf(mrow[r], mx);                                          \
    float a    = exp2f(mrow[r] - mnew);                                       \
    p0v[r] = exp2f(s0 - mnew);                                                \
    p1v[r] = exp2f(s1 - mnew);                                                \
    float rs = p0v[r] + p1v[r];                                               \
    _Pragma("unroll")                                                         \
    for (int off = 1; off < 16; off <<= 1)                                    \
      rs += __shfl_xor(rs, off, 32);                                          \
    lrow[r]  = lrow[r] * a + rs;                                              \
    mrow[r]  = mnew;                                                          \
    alpha[r] = a;                                                             \
  }

// ---------------------------------------------------------------------------
// Kernel 1: flash attention -> z[b][q][h][d] (f16)
// grid.x = B*NH*(S/QTILE); block = 128.  Double-buffered async K/V tiles.
// ---------------------------------------------------------------------------
__global__ __launch_bounds__(128) void attn_flash_z_kernel(
    const float* __restrict__ qg, const f16* __restrict__ kf,
    const f16* __restrict__ vt, f16* __restrict__ zout)
{
  const int bid = blockIdx.x;
  const int qt  = bid % (S_LEN / QTILE);
  const int h   = (bid / (S_LEN / QTILE)) % NH;
  const int b   = bid / ((S_LEN / QTILE) * NH);
  const int q0  = qt * QTILE;

  const int tid  = threadIdx.x;
  const int wid  = __builtin_amdgcn_readfirstlane(tid >> 5);  // scalar wave id
  const int lane = tid & 31;
  const int l16  = lane & 15;
  const int hfl  = lane >> 4;

  __shared__ __align__(16) f16 Klds[2][KTILE][DH + 8];   // buf x [key][d], stride 72
  __shared__ __align__(16) f16 Vlds[2][DH][KTILE + 8];   // buf x [d][key], stride 40
  __shared__ __align__(16) f16 Plds[4][16][KTILE];       // per-wave P tile

  // ---- Q fragments in A-layout (loaded once, f32 source) ----
  const float* qbase =
      qg + ((size_t)(b * S_LEN) + q0 + wid * 16 + l16) * DMODEL + h * DH;
  V16 aq[2];
#pragma unroll
  for (int kc = 0; kc < 2; ++kc)
#pragma unroll
    for (int e = 0; e < 16; ++e) {
      int d = e + 16 * hfl + 32 * kc;
      aq[kc].v[e] = (f16)qbase[d];
    }

  v8f o[4];
#pragma unroll
  for (int nc = 0; nc < 4; ++nc) o[nc] = (v8f){0.f,0.f,0.f,0.f,0.f,0.f,0.f,0.f};
  float mrow[8], lrow[8];
#pragma unroll
  for (int r = 0; r < 8; ++r) { mrow[r] = -1e30f; lrow[r] = 0.f; }

  const int q_row = q0 + wid * 16;           // scalar
  const int nkt   = (q0 + QTILE) / KTILE;

  const uint32_t kbase_lds = lds_addr(&Klds[0][0][0]);
  const uint32_t vbase_lds = lds_addr(&Vlds[0][0][0]);
  const uint32_t KBUFB = KTILE * (DH + 8) * 2;     // 4608 B per K buffer
  const uint32_t VBUFB = DH * (KTILE + 8) * 2;     // 5120 B per V buffer
  const f16* kall = kf + (size_t)(b * S_LEN) * DMODEL + h * DH;
  const f16* vall = vt + (size_t)((b * NH + h) * DH) * S_LEN;

  auto issue_tile = [&](int kt2, int buf2) {
    const int k0b = kt2 * KTILE;
#pragma unroll
    for (int h8 = 0; h8 < 2; ++h8) {
      int c = h8 * 128 + tid;
      int krow = c >> 3, kcol = (c & 7) * 8;
      async_b128(kbase_lds + buf2 * KBUFB + (uint32_t)(krow * (DH + 8) + kcol) * 2,
                 kall + (size_t)(k0b + krow) * DMODEL + kcol);
      int vrow = c >> 2, vcol = (c & 3) * 8;
      async_b128(vbase_lds + buf2 * VBUFB + (uint32_t)(vrow * (KTILE + 8) + vcol) * 2,
                 vall + (size_t)vrow * S_LEN + k0b + vcol);
    }
  };

  issue_tile(0, 0);

  for (int kt = 0; kt < nkt; ++kt) {
    const int k0  = kt * KTILE;
    const int buf = kt & 1;
    if (kt + 1 < nkt) {                 // prefetch next tile into other buffer
      issue_tile(kt + 1, buf ^ 1);
      WAIT_ASYNC(0x4);                  // wait only for tile kt's 4 copies
    } else {
      WAIT_ASYNC(0x0);
    }
    __syncthreads();                    // tile kt visible to all waves

    // scalar, wave-uniform branch: does this wave see any key of this tile?
    if (__builtin_amdgcn_readfirstlane(k0 <= q_row + 15)) {
      // ---- scores ----
      v8f c0 = (v8f){0.f,0.f,0.f,0.f,0.f,0.f,0.f,0.f};
      v8f c1 = c0;
#pragma unroll
      for (int kc = 0; kc < 2; ++kc) {
        V16 bk0, bk1;
        const v8h* p0 = (const v8h*)&Klds[buf][l16][16 * hfl + 32 * kc];
        const v8h* p1 = (const v8h*)&Klds[buf][16 + l16][16 * hfl + 32 * kc];
        bk0.h[0] = p0[0]; bk0.h[1] = p0[1];
        bk1.h[0] = p1[0]; bk1.h[1] = p1[1];
        c0 = __builtin_amdgcn_wmma_f32_16x16x32_f16(false, aq[kc].v, false, bk0.v,
                                                    (short)0, c0, false, false);
        c1 = __builtin_amdgcn_wmma_f32_16x16x32_f16(false, aq[kc].v, false, bk1.v,
                                                    (short)0, c1, false, false);
      }

      // ---- online softmax in exp2 domain; masked only on diagonal tiles ----
      float p0v[8], p1v[8], alpha[8];
      if (__builtin_amdgcn_readfirstlane((k0 + KTILE - 1) > q_row)) {
        SOFTMAX_LOOP(true)
      } else {
        SOFTMAX_LOOP(false)
      }
#pragma unroll
      for (int nc = 0; nc < 4; ++nc)
#pragma unroll
        for (int r = 0; r < 8; ++r) o[nc][r] *= alpha[r];

      // ---- P: C-layout -> per-wave LDS -> A-layout (wave-local ordering) ----
#pragma unroll
      for (int r = 0; r < 8; ++r) {
        Plds[wid][r + 8 * hfl][l16]      = (f16)p0v[r];
        Plds[wid][r + 8 * hfl][16 + l16] = (f16)p1v[r];
      }
      __builtin_amdgcn_wave_barrier();   // wave-private tile; DS is in-order

      V16 ap;
      ap.h[0] = ((const v8h*)&Plds[wid][l16][16 * hfl])[0];
      ap.h[1] = ((const v8h*)&Plds[wid][l16][16 * hfl + 8])[0];

      // ---- PV accumulate ----
#pragma unroll
      for (int nc = 0; nc < 4; ++nc) {
        V16 bv;
        const v8h* vp = (const v8h*)&Vlds[buf][16 * nc + l16][16 * hfl];
        bv.h[0] = vp[0]; bv.h[1] = vp[1];
        o[nc] = __builtin_amdgcn_wmma_f32_16x16x32_f16(false, ap.v, false, bv.v,
                                                       (short)0, o[nc], false, false);
      }
    }
    __syncthreads();   // all waves done with buf before it is re-issued
  }

  // ---- normalize and store z (f16 [b][q][h][d]) ----
#pragma unroll
  for (int nc = 0; nc < 4; ++nc)
#pragma unroll
    for (int r = 0; r < 8; ++r) {
      int qgl = q_row + r + 8 * hfl;
      int d   = 16 * nc + l16;
      zout[(((size_t)b * S_LEN + qgl) * NH + h) * DH + d] =
          (f16)(o[nc][r] / lrow[r]);
    }
}

// ---------------------------------------------------------------------------
// Kernel 2: out[m][n] = z[m][k] * wt[n][k]^T + b_O[n]
// M=4096, K=1024, N=1024.  64x64 block tile, K-step 64, double-buffered async.
// ---------------------------------------------------------------------------
#define KG 64
__global__ __launch_bounds__(128) void out_proj_kernel(
    const f16* __restrict__ z, const f16* __restrict__ wt,
    const float* __restrict__ bias, float* __restrict__ out)
{
  const int n0 = blockIdx.x * 64;
  const int m0 = blockIdx.y * 64;
  const int tid  = threadIdx.x;
  const int wid  = __builtin_amdgcn_readfirstlane(tid >> 5);
  const int lane = tid & 31;
  const int l16  = lane & 15;
  const int hfl  = lane >> 4;

  __shared__ __align__(16) f16 Alds[2][64][KG + 8];   // [m][k]  stride 72
  __shared__ __align__(16) f16 Blds[2][64][KG + 8];   // [n][k]  stride 72

  const uint32_t abase = lds_addr(&Alds[0][0][0]);
  const uint32_t bbase = lds_addr(&Blds[0][0][0]);
  const uint32_t BUFB  = 64 * (KG + 8) * 2;           // 9216 B per buffer

  auto issue_tile = [&](int it, int buf) {
    const int k0 = it * KG;
#pragma unroll
    for (int h8 = 0; h8 < 4; ++h8) {
      int c = h8 * 128 + tid;                // 512 chunks of 8 halves
      int row = c >> 3, col = (c & 7) * 8;   // 64 x 64 halves
      uint32_t loff = (uint32_t)(row * (KG + 8) + col) * 2 + buf * BUFB;
      async_b128(abase + loff, z  + (size_t)(m0 + row) * DMODEL + k0 + col);
      async_b128(bbase + loff, wt + (size_t)(n0 + row) * DMODEL + k0 + col);
    }
  };

  v8f c[4];
#pragma unroll
  for (int nc = 0; nc < 4; ++nc) c[nc] = (v8f){0.f,0.f,0.f,0.f,0.f,0.f,0.f,0.f};

  issue_tile(0, 0);
  for (int it = 0; it < DMODEL / KG; ++it) {
    const int buf = it & 1;
    if (it + 1 < DMODEL / KG) {
      issue_tile(it + 1, buf ^ 1);
      WAIT_ASYNC(0x8);
    } else {
      WAIT_ASYNC(0x0);
    }
    __syncthreads();

#pragma unroll
    for (int kc = 0; kc < 2; ++kc) {
      V16 a;
      a.h[0] = ((const v8h*)&Alds[buf][16 * wid + l16][16 * hfl + 32 * kc])[0];
      a.h[1] = ((const v8h*)&Alds[buf][16 * wid + l16][16 * hfl + 8 + 32 * kc])[0];
#pragma unroll
      for (int nc = 0; nc < 4; ++nc) {
        V16 bf;
        const v8h* bp = (const v8h*)&Blds[buf][16 * nc + l16][16 * hfl + 32 * kc];
        bf.h[0] = bp[0]; bf.h[1] = bp[1];
        c[nc] = __builtin_amdgcn_wmma_f32_16x16x32_f16(false, a.v, false, bf.v,
                                                       (short)0, c[nc], false, false);
      }
    }
    __syncthreads();
  }

#pragma unroll
  for (int nc = 0; nc < 4; ++nc)
#pragma unroll
    for (int r = 0; r < 8; ++r) {
      int m = m0 + 16 * wid + r + 8 * hfl;
      int n = n0 + 16 * nc + l16;
      out[(size_t)m * DMODEL + n] = c[nc][r] + bias[n];
    }
}

// ---------------------------------------------------------------------------
// Kernel 3: residual passthrough
// ---------------------------------------------------------------------------
__global__ void copy_residual_kernel(const float4* __restrict__ src,
                                     float4* __restrict__ dst, int n4)
{
  int i = blockIdx.x * blockDim.x + threadIdx.x;
  if (i < n4) dst[i] = src[i];
}

extern "C" void kernel_launch(void* const* d_in, const int* in_sizes, int n_in,
                              void* d_out, int out_size, void* d_ws, size_t ws_size,
                              hipStream_t stream) {
  const float* residual = (const float*)d_in[0];
  const float* q        = (const float*)d_in[1];
  const float* k        = (const float*)d_in[2];
  const float* v        = (const float*)d_in[3];
  const float* W_O      = (const float*)d_in[4];
  const float* b_O      = (const float*)d_in[5];
  float*       out      = (float*)d_out;

  const int B = 2;
  const size_t BSD = (size_t)B * S_LEN * DMODEL;     // 4,194,304

  // workspace (f16): z | kf | vt | wt  = 8+8+8+2 MB
  f16* z_ws = (f16*)d_ws;
  f16* kf   = z_ws + BSD;
  f16* vt   = kf + BSD;
  f16* wt   = vt + BSD;

  {
    int n4 = (int)(BSD / 4);
    copy_residual_kernel<<<(n4 + 255) / 256, 256, 0, stream>>>(
        (const float4*)residual, (float4*)out, n4);
  }
  convert_kv_kernel<<<B * NH * (S_LEN / 64), 256, 0, stream>>>(k, v, kf, vt);
  convert_w_kernel<<<dim3(DMODEL / 64, DMODEL / 64), 256, 0, stream>>>(W_O, wt);
  attn_flash_z_kernel<<<B * NH * (S_LEN / QTILE), 128, 0, stream>>>(q, kf, vt, z_ws);
  out_proj_kernel<<<dim3(DMODEL / 64, (B * S_LEN) / 64), 128, 0, stream>>>(
      z_ws, wt, b_O, out + BSD);
}
